// VectorAttention_63153199120693
// MI455X (gfx1250) — matrix operand order repaired and checked by reading, not compiled, not run on hardware
//
#include <hip/hip_runtime.h>
#include <hip/hip_bf16.h>

#define B_ 2
#define N_ 512
#define C_ 128

typedef __attribute__((ext_vector_type(16))) _Float16     v16h;
typedef __attribute__((ext_vector_type(8)))  _Float16     v8h;
typedef __attribute__((ext_vector_type(8)))  float        v8f;
typedef __attribute__((ext_vector_type(4)))  unsigned int v4u;
typedef __attribute__((ext_vector_type(8)))  int          v8i;
typedef __attribute__((ext_vector_type(4)))  int          v4i;

__device__ __forceinline__ float mish_f(float x) {
    // mish(x) = x * tanh(softplus(x)) ; tanh(log(y)) = (y^2-1)/(y^2+1), y = 1+e^x
    float e  = __expf(fminf(x, 30.0f));
    float y  = 1.0f + e;
    float y2 = y * y;
    return x * ((y2 - 1.0f) / (y2 + 1.0f));
}

__device__ __forceinline__ v8f wmma16(v16h a, v16h b, v8f c) {
    return __builtin_amdgcn_wmma_f32_16x16x32_f16(
        /*neg_a=*/false, a, /*neg_b=*/false, b,
        /*c_mod=*/(short)0, c, /*reuse_a=*/false, /*reuse_b=*/false);
}

// ---------------------------------------------------------------------------
// TDM flat copy: nbytes (2-byte elements) from global -> LDS byte offset.
// D# per CDNA5 ISA: group0 = {count|flags,
// lds_addr, global_addr lo, global_addr hi | type=2}; group1 = data_size=2B,
// tensor_dim0 = tile_dim0 = nelem, tensor_dim1 = tile_dim1 = 1.
// ---------------------------------------------------------------------------
__device__ __forceinline__ void tdm_load_to_lds(unsigned lds_off, const void* gsrc,
                                                unsigned nbytes) {
#if __has_builtin(__builtin_amdgcn_tensor_load_to_lds)
    const unsigned long long ga = (unsigned long long)gsrc;
    const unsigned nelem = nbytes >> 1;              // 2-byte elements (<= 0xFFFF)
    v4u g0;
    g0[0] = 1u;                                       // count=1 (valid), no gather
    g0[1] = lds_off;                                  // lds_addr (bytes)
    g0[2] = (unsigned)(ga & 0xffffffffu);             // global_addr[31:0]
    g0[3] = ((unsigned)(ga >> 32) & 0x1ffffffu) | (2u << 30);  // addr[56:32] | type=2
    v8i g1;
    g1[0] = (int)(1u << 16);                          // workgroup_mask=0, data_size=1 (2B)
    g1[1] = (int)((nelem & 0xffffu) << 16);           // tensor_dim0[15:0]
    g1[2] = (int)((nelem >> 16) | (1u << 16));        // tensor_dim0[31:16], tensor_dim1=1
    g1[3] = (int)((nelem & 0xffffu) << 16);           // tensor_dim1 hi=0, tile_dim0=nelem
    g1[4] = 1;                                        // tile_dim1=1, tile_dim2=0
    g1[5] = (int)nelem;                               // tensor_dim0_stride[31:0]
    g1[6] = 0;
    g1[7] = 0;
    v4i z4 = {0, 0, 0, 0};
#if __clang_major__ >= 23
    v8i z8 = {0, 0, 0, 0, 0, 0, 0, 0};
    __builtin_amdgcn_tensor_load_to_lds(g0, g1, z4, z4, z8, 0);
#else
    __builtin_amdgcn_tensor_load_to_lds(g0, g1, z4, z4, 0);
#endif
#endif
}

// ---------------------------------------------------------------------------
// Tiny prep: convert W1/W2 to f16 once (global), so phase-2 TDM is a raw copy.
// ---------------------------------------------------------------------------
__global__ __launch_bounds__(256) void cvtw_kernel(const float* __restrict__ W1,
                                                   const float* __restrict__ W2,
                                                   _Float16* __restrict__ w1h,
                                                   _Float16* __restrict__ w2h) {
    const int idx = blockIdx.x * 256 + threadIdx.x;
    if (idx < C_ * C_) {
        w1h[idx] = (_Float16)W1[idx];
        w2h[idx] = (_Float16)W2[idx];
    }
}

// ---------------------------------------------------------------------------
// Phase 1: q/k/v projections.  One wave handles 16 rows of feat.
//   A layout (16-bit, 16x32): lane m=L%16, half=L/16; elem h -> K = kc*32 +
//   (h/8)*16 + half*8 + (h%8).
//   B layout (16-bit, 32x16): lane n=L%16; elem h -> K = kc*32 + (h/8)*16 + half*8 + (h%8).
//   C layout (f32 16x16): lane n=L%16; reg r -> row m = r + 8*half.
// q/k stored as f16 (post LN+mish, O(1) values); v stored f32.
// ---------------------------------------------------------------------------
template <bool DO_LN, typename T>
__device__ __forceinline__ void project16(
    const v16h A[4], const float* __restrict__ W,
    const float* __restrict__ bias, const float* __restrict__ gam,
    const float* __restrict__ bet, T* __restrict__ outrow0, int lane)
{
    const int l16  = lane & 15;
    const int half = lane >> 4;
    v8f t[8];
#pragma unroll
    for (int nt = 0; nt < 8; ++nt) {
        v8f c = {};
#pragma unroll
        for (int kc = 0; kc < 4; ++kc) {
            const float* wr = W + (size_t)(nt * 16 + l16) * C_ + kc * 32 + half * 8;
            v16h Bv;
#pragma unroll
            for (int t4 = 0; t4 < 4; ++t4) {
                float4 x = *(const float4*)(wr + (t4 >> 1) * 16 + (t4 & 1) * 4);
                Bv[t4 * 4 + 0] = (_Float16)x.x;
                Bv[t4 * 4 + 1] = (_Float16)x.y;
                Bv[t4 * 4 + 2] = (_Float16)x.z;
                Bv[t4 * 4 + 3] = (_Float16)x.w;
            }
            c = wmma16(A[kc], Bv, c);
        }
        float bc = bias[nt * 16 + l16];
#pragma unroll
        for (int r = 0; r < 8; ++r) c[r] += bc;
        t[nt] = c;
    }
    if (DO_LN) {
#pragma unroll
        for (int r = 0; r < 8; ++r) {
            float s1 = 0.f, s2 = 0.f;
#pragma unroll
            for (int nt = 0; nt < 8; ++nt) { float x = t[nt][r]; s1 += x; s2 += x * x; }
#pragma unroll
            for (int off = 1; off < 16; off <<= 1) {
                s1 += __shfl_xor(s1, off);
                s2 += __shfl_xor(s2, off);
            }
            float mu   = s1 * (1.0f / 128.0f);
            float var  = s2 * (1.0f / 128.0f) - mu * mu;
            float rstd = rsqrtf(var + 1e-5f);
#pragma unroll
            for (int nt = 0; nt < 8; ++nt) {
                float x = (t[nt][r] - mu) * rstd * gam[nt * 16 + l16] + bet[nt * 16 + l16];
                t[nt][r] = mish_f(x);
            }
        }
    }
#pragma unroll
    for (int nt = 0; nt < 8; ++nt)
#pragma unroll
        for (int r = 0; r < 8; ++r)
            outrow0[(size_t)(r + 8 * half) * C_ + nt * 16 + l16] = (T)t[nt][r];
}

__global__ __launch_bounds__(32) void qkv_kernel(
    const float* __restrict__ feat,
    const float* __restrict__ Wq, const float* __restrict__ bq,
    const float* __restrict__ gq, const float* __restrict__ betaq,
    const float* __restrict__ Wk, const float* __restrict__ bk,
    const float* __restrict__ gk, const float* __restrict__ betak,
    const float* __restrict__ Wv, const float* __restrict__ bv,
    _Float16* __restrict__ qh, _Float16* __restrict__ kh, float* __restrict__ vb)
{
    const int lane = threadIdx.x;
    const int l16  = lane & 15;
    const int half = lane >> 4;
    const int row0 = blockIdx.x * 16;   // over B*N = 1024 rows

    v16h A[4];
    const float* frow = feat + (size_t)(row0 + l16) * C_;
#pragma unroll
    for (int kc = 0; kc < 4; ++kc) {
#pragma unroll
        for (int g = 0; g < 2; ++g) {
            const int base = kc * 32 + g * 16 + half * 8;
            float4 x0 = *(const float4*)(frow + base);
            float4 x1 = *(const float4*)(frow + base + 4);
            A[kc][g * 8 + 0] = (_Float16)x0.x;
            A[kc][g * 8 + 1] = (_Float16)x0.y;
            A[kc][g * 8 + 2] = (_Float16)x0.z;
            A[kc][g * 8 + 3] = (_Float16)x0.w;
            A[kc][g * 8 + 4] = (_Float16)x1.x;
            A[kc][g * 8 + 5] = (_Float16)x1.y;
            A[kc][g * 8 + 6] = (_Float16)x1.z;
            A[kc][g * 8 + 7] = (_Float16)x1.w;
        }
    }
    project16<true,  _Float16>(A, Wq, bq, gq, betaq, qh + (size_t)row0 * C_, lane);
    project16<true,  _Float16>(A, Wk, bk, gk, betak, kh + (size_t)row0 * C_, lane);
    project16<false, float   >(A, Wv, bv, nullptr, nullptr, vb + (size_t)row0 * C_, lane);
}

// ---------------------------------------------------------------------------
// Phase 2: fused rel-MLP + per-channel online softmax + weighted sum of v.
// 4 waves/block; each wave owns one (b,i); W1h/W2h pulled into LDS via TDM.
// ---------------------------------------------------------------------------
__global__ __launch_bounds__(128) void attn_kernel(
    const _Float16* __restrict__ qh, const _Float16* __restrict__ kh,
    const float* __restrict__ vb, const float* __restrict__ dist,
    const _Float16* __restrict__ w1h, const float* __restrict__ b1,
    const float* __restrict__ g1, const float* __restrict__ be1,
    const _Float16* __restrict__ w2h, const float* __restrict__ b2,
    float* __restrict__ out)
{
    extern __shared__ __align__(16) _Float16 smem[];
    _Float16* W1s = smem;                 // 128*128 halves @ LDS byte offset 0
    _Float16* W2s = smem + C_ * C_;       // 128*128 halves @ LDS byte offset 32768
    _Float16* A2  = smem + 2 * C_ * C_;   // 4 waves * 16*128 halves @ 65536

    const int tid = threadIdx.x;
#if __has_builtin(__builtin_amdgcn_tensor_load_to_lds)
    if (tid < 32) {                       // uniform branch: only wave 0 issues TDM
        tdm_load_to_lds(0u,     w1h, C_ * C_ * 2);
        tdm_load_to_lds(32768u, w2h, C_ * C_ * 2);
        __builtin_amdgcn_s_wait_tensorcnt(0);
    }
#else
    for (int idx = tid; idx < C_ * C_; idx += 128) {
        W1s[idx] = w1h[idx];
        W2s[idx] = w2h[idx];
    }
#endif
    __syncthreads();

    const int wave = tid >> 5;
    const int lane = tid & 31;
    const int l16  = lane & 15;
    const int half = lane >> 4;
    const int gi   = blockIdx.x * 4 + wave;   // over B*N
    const int b    = gi / N_;
    const int i    = gi % N_;
    _Float16* a2w  = A2 + wave * (16 * C_);

    float b1c[8], g1c[8], be1c[8], b2c[8];
#pragma unroll
    for (int nt = 0; nt < 8; ++nt) {
        const int c = nt * 16 + l16;
        b1c[nt] = b1[c]; g1c[nt] = g1[c]; be1c[nt] = be1[c]; b2c[nt] = b2[c];
    }
    const float di = dist[b * N_ + i];
    const _Float16* qrow = qh + (size_t)(b * N_ + i) * C_;

    float mrun[8], srun[8], arun[8];
#pragma unroll
    for (int nt = 0; nt < 8; ++nt) { mrun[nt] = -3.0e38f; srun[nt] = 0.f; arun[nt] = 0.f; }

    for (int jt = 0; jt < N_ / 16; ++jt) {
        // ---- build rel = k[j] - q[i] in A layout, packed f16 math ----
        v16h A[4];
        const _Float16* krow = kh + (size_t)(b * N_ + jt * 16 + l16) * C_;
#pragma unroll
        for (int kc = 0; kc < 4; ++kc) {
#pragma unroll
            for (int g = 0; g < 2; ++g) {
                const int base = kc * 32 + g * 16 + half * 8;
                v8h kd = *(const v8h*)(krow + base);
                v8h qd = *(const v8h*)(qrow + base);
                v8h rd = kd - qd;                     // v_pk_add_f16 (sub)
#pragma unroll
                for (int h = 0; h < 8; ++h) A[kc][g * 8 + h] = rd[h];
            }
        }
        // ---- GEMM1: t = rel @ W1^T + b1 ----
        v8f t[8];
#pragma unroll
        for (int nt = 0; nt < 8; ++nt) {
            v8f c = {};
#pragma unroll
            for (int kc = 0; kc < 4; ++kc) {
                const _Float16* wr = W1s + (nt * 16 + l16) * C_ + kc * 32 + half * 8;
                v8h blo = *(const v8h*)(wr);
                v8h bhi = *(const v8h*)(wr + 16);
                v16h Bv;
#pragma unroll
                for (int h = 0; h < 8; ++h) { Bv[h] = blo[h]; Bv[8 + h] = bhi[h]; }
                c = wmma16(A[kc], Bv, c);
            }
#pragma unroll
            for (int r = 0; r < 8; ++r) c[r] += b1c[nt];
            t[nt] = c;
        }
        // ---- LayerNorm over channels (per row) + mish ----
#pragma unroll
        for (int r = 0; r < 8; ++r) {
            float s1 = 0.f, s2 = 0.f;
#pragma unroll
            for (int nt = 0; nt < 8; ++nt) { float x = t[nt][r]; s1 += x; s2 += x * x; }
#pragma unroll
            for (int off = 1; off < 16; off <<= 1) {
                s1 += __shfl_xor(s1, off);
                s2 += __shfl_xor(s2, off);
            }
            float mu   = s1 * (1.0f / 128.0f);
            float var  = s2 * (1.0f / 128.0f) - mu * mu;
            float rstd = rsqrtf(var + 1e-5f);
#pragma unroll
            for (int nt = 0; nt < 8; ++nt) {
                float x = (t[nt][r] - mu) * rstd * g1c[nt] + be1c[nt];
                t[nt][r] = mish_f(x);
            }
        }
        // ---- re-layout C -> A via per-wave LDS scratch (same-wave DS order) ----
#pragma unroll
        for (int nt = 0; nt < 8; ++nt)
#pragma unroll
            for (int r = 0; r < 8; ++r)
                a2w[(r + 8 * half) * C_ + nt * 16 + l16] = (_Float16)t[nt][r];

        v16h A2v[4];
#pragma unroll
        for (int kc = 0; kc < 4; ++kc) {
            v8h lo = *(const v8h*)(a2w + l16 * C_ + kc * 32 + half * 8);
            v8h hi = *(const v8h*)(a2w + l16 * C_ + kc * 32 + 16 + half * 8);
#pragma unroll
            for (int h = 0; h < 8; ++h) { A2v[kc][h] = lo[h]; A2v[kc][8 + h] = hi[h]; }
        }
        // ---- GEMM2: w = h @ W2^T + b2 (reuse t) ----
#pragma unroll
        for (int nt = 0; nt < 8; ++nt) {
            v8f c = {};
#pragma unroll
            for (int kc = 0; kc < 4; ++kc) {
                const _Float16* wr = W2s + (nt * 16 + l16) * C_ + kc * 32 + half * 8;
                v8h blo = *(const v8h*)(wr);
                v8h bhi = *(const v8h*)(wr + 16);
                v16h Bv;
#pragma unroll
                for (int h = 0; h < 8; ++h) { Bv[h] = blo[h]; Bv[8 + h] = bhi[h]; }
                c = wmma16(A2v[kc], Bv, c);
            }
#pragma unroll
            for (int r = 0; r < 8; ++r) c[r] += b2c[nt];
            t[nt] = c;
        }
        // ---- online per-channel softmax update ----
#pragma unroll
        for (int nt = 0; nt < 8; ++nt) {
            float mx = t[nt][0];
#pragma unroll
            for (int r = 1; r < 8; ++r) mx = fmaxf(mx, t[nt][r]);
            mx = fmaxf(mx, __shfl_xor(mx, 16));   // combine lane halves (rows 0-7 | 8-15)
            float nm = fmaxf(mrun[nt], mx);
            float sc = __expf(mrun[nt] - nm);
            srun[nt] *= sc;
            arun[nt] *= sc;
            mrun[nt] = nm;
        }
#pragma unroll
        for (int r = 0; r < 8; ++r) {
            const int   j  = jt * 16 + r + 8 * half;
            const float dj = dist[b * N_ + j];
            const float mk = (di * dj > 0.f) ? 1.f : 0.f;
            const float* vrow = vb + (size_t)(b * N_ + j) * C_;
#pragma unroll
            for (int nt = 0; nt < 8; ++nt) {
                float e = __expf(t[nt][r] - mrun[nt]) * dj;
                srun[nt] += e;                                  // denominator: no mask
                arun[nt] += e * mk * vrow[nt * 16 + l16];       // numerator: masked
            }
        }
    }
    // ---- combine halves and write out[b,i,:] ----
#pragma unroll
    for (int nt = 0; nt < 8; ++nt) {
        float st = srun[nt] + __shfl_xor(srun[nt], 16);
        float at = arun[nt] + __shfl_xor(arun[nt], 16);
        if (half == 0)
            out[(size_t)(b * N_ + i) * C_ + nt * 16 + l16] = at / st;
    }
}

extern "C" void kernel_launch(void* const* d_in, const int* in_sizes, int n_in,
                              void* d_out, int out_size, void* d_ws, size_t ws_size,
                              hipStream_t stream) {
    const float* feat  = (const float*)d_in[0];
    /* d_in[1] = pos : unused by reference */
    const float* dist  = (const float*)d_in[2];
    const float* Wq    = (const float*)d_in[3];
    const float* bq    = (const float*)d_in[4];
    const float* gq    = (const float*)d_in[5];
    const float* betaq = (const float*)d_in[6];
    const float* Wk    = (const float*)d_in[7];
    const float* bk    = (const float*)d_in[8];
    const float* gk    = (const float*)d_in[9];
    const float* betak = (const float*)d_in[10];
    const float* Wv    = (const float*)d_in[11];
    const float* bv    = (const float*)d_in[12];
    const float* W1    = (const float*)d_in[13];
    const float* b1    = (const float*)d_in[14];
    const float* g1    = (const float*)d_in[15];
    const float* be1   = (const float*)d_in[16];
    const float* W2    = (const float*)d_in[17];
    const float* b2    = (const float*)d_in[18];
    float* out = (float*)d_out;

    // workspace layout
    _Float16* qh  = (_Float16*)d_ws;                       // B*N*C f16
    _Float16* kh  = qh + (size_t)B_ * N_ * C_;             // B*N*C f16
    float*    vbw = (float*)(kh + (size_t)B_ * N_ * C_);   // B*N*C f32
    _Float16* w1h = (_Float16*)(vbw + (size_t)B_ * N_ * C_); // C*C f16
    _Float16* w2h = w1h + (size_t)C_ * C_;                   // C*C f16

    cvtw_kernel<<<(C_ * C_ + 255) / 256, 256, 0, stream>>>(W1, W2, w1h, w2h);

    qkv_kernel<<<(B_ * N_) / 16, 32, 0, stream>>>(
        feat, Wq, bq, gq, betaq, Wk, bk, gk, betak, Wv, bv, qh, kh, vbw);

    const size_t shmem = (size_t)(2 * C_ * C_ + 4 * 16 * C_) * sizeof(_Float16); // 80 KB
    (void)hipFuncSetAttribute((const void*)attn_kernel,
                              hipFuncAttributeMaxDynamicSharedMemorySize,
                              (int)shmem);
    attn_kernel<<<(B_ * N_) / 4, 128, shmem, stream>>>(
        qh, kh, vbw, dist, w1h, b1, g1, be1, w2h, b2, out);
}
